// Model_82437602279966
// MI455X (gfx1250) — matrix operand order repaired and checked by reading, not compiled, not run on hardware
//
#include <hip/hip_runtime.h>
#include <hip/hip_bf16.h>

// ---------------------------------------------------------------------------
// MI455X (gfx1250, wave32) sparse-CNN forward.
// Matmuls: V_WMMA_F32_16X16X32_BF16 (bf16 operands, fp32 accumulate),
// all A/B fragments loaded as aligned b128 vectors (K-contiguous layout +
// weights transposed to [Cout, K] at convert time).
// BN statistics: fp32 LDS atomics -> global atomics (two-pass, masked sites).
// ---------------------------------------------------------------------------

typedef __attribute__((ext_vector_type(16))) __bf16 v16bf;
typedef __attribute__((ext_vector_type(8)))  __bf16 v8bf;
typedef __attribute__((ext_vector_type(8)))  float  v8f;

#define BN_EPS 1e-4f

// ------------------------------ utility kernels ----------------------------

__global__ void k_zero(float* __restrict__ p, int n) {
  int i = blockIdx.x * blockDim.x + threadIdx.x;
  if (i < n) p[i] = 0.f;
}

// convert + transpose weights: w[K,Cout] fp32  ->  wT[Cout,K] bf16
__global__ void k_cvt_wT(const float* __restrict__ w, __bf16* __restrict__ wT,
                         int K, int Cout) {
  int i = blockIdx.x * blockDim.x + threadIdx.x;
  if (i >= K * Cout) return;
  int k = i / Cout, co = i - k * Cout;
  wT[(long)co * K + k] = (__bf16)w[i];
}

// any-active pooling of a float mask over a 3x3 window (VALID)
__global__ void k_pool_mask(const float* __restrict__ mi, float* __restrict__ mo,
                            int B, int Hi, int Wi, int Ho, int Wo, int stride) {
  int i = blockIdx.x * blockDim.x + threadIdx.x;
  int total = B * Ho * Wo;
  if (i >= total) return;
  int b = i / (Ho * Wo), r = i % (Ho * Wo), oy = r / Wo, ox = r % Wo;
  float any = 0.f;
  for (int dy = 0; dy < 3; ++dy)
    for (int dx = 0; dx < 3; ++dx) {
      int iy = oy * stride + dy, ix = ox * stride + dx;
      if (iy < Hi && ix < Wi && mi[((long)b * Hi + iy) * Wi + ix] > 0.f) any = 1.f;
    }
  mo[i] = any;
}

// Fused: submanifold conv 3->16 (3x3 SAME, masked in/out) + sparse maxpool 3x3/s2
// over active sites only. 63x63 -> 31x31. Avoids materializing 63x63x16 (260MB).
__global__ void k_conv0_pool(const float* __restrict__ feat,
                             const unsigned char* __restrict__ mk,
                             const float* __restrict__ w0,   // [3,3,3,16] fp32
                             float* __restrict__ xo,         // [B,31,31,16]
                             float* __restrict__ mo,         // [B,31,31]
                             int B) {
  int i = blockIdx.x * blockDim.x + threadIdx.x;
  int total = B * 31 * 31;
  if (i >= total) return;
  int b = i / (31 * 31), r = i % (31 * 31), oy = r / 31, ox = r % 31;
  float best[16];
  bool any = false;
  for (int py = 0; py < 3; ++py)
    for (int px = 0; px < 3; ++px) {
      int iy = oy * 2 + py, ix = ox * 2 + px;           // 0..62, always in-bounds
      if (!mk[((long)b * 63 + iy) * 63 + ix]) continue; // pool over ACTIVE sites only
      float acc[16];
#pragma unroll
      for (int c = 0; c < 16; ++c) acc[c] = 0.f;
      for (int dy = 0; dy < 3; ++dy) {
        int yy = iy + dy - 1;
        if ((unsigned)yy >= 63u) continue;
        for (int dx = 0; dx < 3; ++dx) {
          int xx = ix + dx - 1;
          if ((unsigned)xx >= 63u) continue;
          if (!mk[((long)b * 63 + yy) * 63 + xx]) continue; // input pre-masked
          const float* fp = feat + (((long)b * 63 + yy) * 63 + xx) * 3;
          const float* wp = w0 + ((dy * 3 + dx) * 3) * 16;
#pragma unroll
          for (int ci = 0; ci < 3; ++ci) {
            float f = fp[ci];
#pragma unroll
            for (int c = 0; c < 16; ++c) acc[c] = fmaf(f, wp[ci * 16 + c], acc[c]);
          }
        }
      }
      if (!any) {
        any = true;
#pragma unroll
        for (int c = 0; c < 16; ++c) best[c] = acc[c];
      } else {
#pragma unroll
        for (int c = 0; c < 16; ++c) best[c] = fmaxf(best[c], acc[c]);
      }
    }
  mo[i] = any ? 1.f : 0.f;
#pragma unroll
  for (int c = 0; c < 16; ++c) xo[(long)i * 16 + c] = any ? best[c] : 0.f;
}

// ------------------------------ BatchNormReLU ------------------------------
// stats layout: [0]=active count, [1..C]=sum, [1+C..2C]=sumsq  (active sites)

__global__ void k_bn_stats(const float* __restrict__ x, const float* __restrict__ mask,
                           float* __restrict__ stats, int nSites, int C) {
  __shared__ float sh[257];
  int t = threadIdx.x;
  for (int i = t; i < 2 * C + 1; i += blockDim.x) sh[i] = 0.f;
  __syncthreads();
  long total = (long)nSites * C;
  for (long i = (long)blockIdx.x * blockDim.x + t; i < total;
       i += (long)gridDim.x * blockDim.x) {
    int c = (int)(i % C);
    int s = (int)(i / C);
    if (mask[s] > 0.f) {
      float v = x[i];
      atomicAdd(&sh[1 + c], v);
      atomicAdd(&sh[1 + C + c], v * v);
      if (c == 0) atomicAdd(&sh[0], 1.f);
    }
  }
  __syncthreads();
  for (int i = t; i < 2 * C + 1; i += blockDim.x) atomicAdd(&stats[i], sh[i]);
}

__global__ void k_bn_apply(const float* __restrict__ x, const float* __restrict__ mask,
                           const float* __restrict__ stats,
                           const float* __restrict__ gamma, const float* __restrict__ beta,
                           float* __restrict__ y32, __bf16* __restrict__ ybf,
                           int nSites, int C) {
  long i = (long)blockIdx.x * blockDim.x + threadIdx.x;
  long total = (long)nSites * C;
  if (i >= total) return;
  int c = (int)(i % C);
  int s = (int)(i / C);
  float cnt  = fmaxf(stats[0], 1.f);
  float mean = stats[1 + c] / cnt;
  float var  = stats[1 + C + c] / cnt - mean * mean;  // E[x^2]-mean^2 over active
  float v = (x[i] - mean) * rsqrtf(var + BN_EPS) * gamma[c] + beta[c];
  v = fmaxf(v, 0.f) * mask[s];
  if (y32) y32[i] = v;
  ybf[i] = (__bf16)v;
}

// ------------------------- implicit-GEMM 3x3 conv --------------------------
// Templated on <Cin,Cout,STRIDE,PAD>: all tap decodes fold at compile time.
// A fragment (ISA 16-bit A 16x32 layout): each 8-element K-run is contiguous
// (one pixel, consecutive ci, Cin%8==0) -> one aligned b128 load per run.
// B fragment (32x16 column-striped): weights pre-transposed to [Cout,Ktot]
// -> per lane two 8-wide K runs at k0+8*half and k0+16+8*half -> two aligned b128 loads.
// Ktot = 9*Cin is a multiple of 16, so runs are fully valid or fully zero.

template <int Cin, int STRIDE, int PAD>
__device__ inline v8bf load_a_run(const __bf16* __restrict__ inB, int s,
                                  int oy, int ox, int Hi, int Wi, bool mv) {
  constexpr int Ktot = 9 * Cin;
  v8bf r = {};
  if (mv && s < Ktot) {
    int f  = s / Cin;             // compile-time strength-reduced
    int ci = s - f * Cin;
    int dy = f / 3, dx = f - dy * 3;
    int iy = oy * STRIDE + dy - PAD;
    int ix = ox * STRIDE + dx - PAD;
    if ((unsigned)iy < (unsigned)Hi && (unsigned)ix < (unsigned)Wi)
      r = *(const v8bf*)(inB + ((long)iy * Wi + ix) * Cin + ci);
  }
  return r;
}

template <int Cin, int Cout, int STRIDE, int PAD>
__global__ void __launch_bounds__(128)
k_conv3x3_wmma(const __bf16* __restrict__ in, const __bf16* __restrict__ wT,
               const float* __restrict__ maskO, const float* __restrict__ res,
               float* __restrict__ out, int B, int Hi, int Wi, int Ho, int Wo) {
  constexpr int Ktot = 9 * Cin;
  const int lane = threadIdx.x;
  const int row  = lane & 15;
  const int half = lane >> 4;
  const int M    = B * Ho * Wo;
  const int mTile = blockIdx.x * 4 + threadIdx.y;   // 4 waves, one M-tile each
  int m = mTile * 16 + row;
  bool mv = m < M;
  int mc = mv ? m : 0;
  int b  = mc / (Ho * Wo);
  int r  = mc % (Ho * Wo);
  int oy = r / Wo, ox = r % Wo;
  const int ncol = blockIdx.y * 16 + row;           // Cout is a multiple of 16
  const __bf16* inB  = in + (long)b * Hi * Wi * Cin;
  const __bf16* wrow = wT + (long)ncol * Ktot;
  if (mv)
    __builtin_prefetch(inB + ((long)(oy * STRIDE) * Wi + ox * STRIDE) * Cin, 0, 3);
  v8f acc = {};
#pragma unroll
  for (int k0 = 0; k0 < Ktot; k0 += 32) {
    v8bf alo = load_a_run<Cin, STRIDE, PAD>(inB, k0 + half * 8, oy, ox, Hi, Wi, mv);
    v8bf ahi = load_a_run<Cin, STRIDE, PAD>(inB, k0 + 16 + half * 8, oy, ox, Hi, Wi, mv);
    v16bf a = __builtin_shufflevector(alo, ahi, 0, 1, 2, 3, 4, 5, 6, 7,
                                      8, 9, 10, 11, 12, 13, 14, 15);
    v16bf bw = {};
    int kb = k0 + half * 8;
    {
      v8bf b0 = *(const v8bf*)(wrow + kb);
      v8bf b1 = {};
      if (kb + 16 < Ktot) b1 = *(const v8bf*)(wrow + kb + 16);
      bw = __builtin_shufflevector(b0, b1, 0, 1, 2, 3, 4, 5, 6, 7,
                                   8, 9, 10, 11, 12, 13, 14, 15);
    }
    acc = __builtin_amdgcn_wmma_f32_16x16x32_bf16(false, a, false, bw,
                                                  (short)0, acc, false, false);
  }
#pragma unroll
  for (int j = 0; j < 8; ++j) {
    int mm = mTile * 16 + j + 8 * half;             // D layout: VGPR j -> M=j / j+8
    if (mm < M) {
      float v = acc[j] * maskO[mm];
      if (res) v += res[(long)mm * Cout + ncol];
      out[(long)mm * Cout + ncol] = v;
    }
  }
}

template <int Cin, int Cout, int STRIDE, int PAD>
static void run_conv(const __bf16* in, const float* w_f32, __bf16* wscr,
                     const float* maskO, const float* res, float* out_,
                     int B, int Hi, int Wi, int Ho, int Wo, hipStream_t stream) {
  constexpr int K = 9 * Cin;
  int n = K * Cout;
  k_cvt_wT<<<(n + 255) / 256, 256, 0, stream>>>(w_f32, wscr, K, Cout);
  int M = B * Ho * Wo;
  int mT = (M + 15) / 16;
  dim3 grid((mT + 3) / 4, Cout / 16);
  dim3 blk(32, 4);
  k_conv3x3_wmma<Cin, Cout, STRIDE, PAD>
      <<<grid, blk, 0, stream>>>(in, wscr, maskO, res, out_, B, Hi, Wi, Ho, Wo);
}

// ------------------------------ classifier GEMM ----------------------------
// out[M,N] = A[M,K] @ B[K,N] + bias, B pre-transposed to BT[N,K].
// All fragment loads are aligned b128 vectors (K=128).

__global__ void __launch_bounds__(128)
k_gemm_wmma(const __bf16* __restrict__ A, const __bf16* __restrict__ BT,
            const float* __restrict__ bias, float* __restrict__ out,
            int M, int K, int N) {
  const int lane = threadIdx.x;
  const int row  = lane & 15;
  const int half = lane >> 4;
  const int mTile = blockIdx.x * 4 + threadIdx.y;
  int m = mTile * 16 + row;
  int ncol = blockIdx.y * 16 + row;
  bool mv = m < M, nv = ncol < N;
  const __bf16* arow = A  + (long)(mv ? m : 0) * K;
  const __bf16* brow = BT + (long)(nv ? ncol : 0) * K;
  v8f acc = {};
  for (int k0 = 0; k0 < K; k0 += 32) {
    v16bf a = {}, bw = {};
    if (mv) {
      v8bf lo = *(const v8bf*)(arow + k0 + half * 8);
      v8bf hi = *(const v8bf*)(arow + k0 + 16 + half * 8);
      a = __builtin_shufflevector(lo, hi, 0, 1, 2, 3, 4, 5, 6, 7,
                                  8, 9, 10, 11, 12, 13, 14, 15);
    }
    if (nv) {
      int kb = k0 + half * 8;
      v8bf b0 = *(const v8bf*)(brow + kb);
      v8bf b1 = *(const v8bf*)(brow + kb + 16);
      bw = __builtin_shufflevector(b0, b1, 0, 1, 2, 3, 4, 5, 6, 7,
                                   8, 9, 10, 11, 12, 13, 14, 15);
    }
    acc = __builtin_amdgcn_wmma_f32_16x16x32_bf16(false, a, false, bw,
                                                  (short)0, acc, false, false);
  }
#pragma unroll
  for (int j = 0; j < 8; ++j) {
    int mm = mTile * 16 + j + 8 * half;
    if (mm < M && ncol < N)
      out[(long)mm * N + ncol] = acc[j] + bias[ncol];
  }
}

// ------------------------------- orchestration -----------------------------

extern "C" void kernel_launch(void* const* d_in, const int* in_sizes, int n_in,
                              void* d_out, int out_size, void* d_ws, size_t ws_size,
                              hipStream_t stream) {
  (void)in_sizes; (void)n_in; (void)out_size; (void)ws_size;
  const int B = 1024;

  auto F = [&](int i) { return (const float*)d_in[i]; };

  // JAX pytree flatten (dict keys sorted): features, mask, then params leaves:
  // bn_last_b, bn_last_g, bnf_b, bnf_g, groups[ (p0,p1) x4 ], lin_b, lin_w, w0, wf
  // per p-dict sorted keys: bn0_b, bn0_g, bn1_b, bn1_g, wa, wb, (wid)
  const float*         features = F(0);
  const unsigned char* maskIn   = (const unsigned char*)d_in[1];
  const float* bn_last_b = F(2);
  const float* bn_last_g = F(3);
  const float* bnf_b = F(4);
  const float* bnf_g = F(5);
  const int g0p0 = 6,  g0p1 = 12;                 // group0: 16->16, s1  (6+6 leaves)
  const int g1p0 = 18, g1p1 = 25;                 // group1: 16->32, s2  (7+6)
  const int g2p0 = 31, g2p1 = 38;                 // group2: 32->48, s2  (7+6)
  const int g3p0 = 44, g3p1 = 51;                 // group3: 48->96, s2  (7+6)
  const float* lin_b = F(57);
  const float* lin_w = F(58);                     // [128,3755]
  const float* w0    = F(59);                     // [3,3,3,16]
  const float* wf    = F(60);                     // [3,3,96,128]

  // ---- workspace bump allocator (total ~380 MB) ----
  char*  base = (char*)d_ws;
  size_t off  = 0;
  auto alloc = [&](size_t bytes) -> void* {
    void* p = base + off;
    off += (bytes + 255) & ~(size_t)255;
    return p;
  };
  const int s31 = B * 31 * 31, s15 = B * 15 * 15, s7 = B * 7 * 7, s3 = B * 3 * 3, sF = B;

  float* M31 = (float*)alloc((size_t)s31 * 4);
  float* M15 = (float*)alloc((size_t)s15 * 4);
  float* M7  = (float*)alloc((size_t)s7  * 4);
  float* M3  = (float*)alloc((size_t)s3  * 4);
  float* MF  = (float*)alloc((size_t)sF  * 4);

  float*  XA31 = (float*)alloc((size_t)s31 * 16 * 4);
  float*  XB31 = (float*)alloc((size_t)s31 * 16 * 4);
  float*  H31  = (float*)alloc((size_t)s31 * 16 * 4);
  __bf16* Y31  = (__bf16*)alloc((size_t)s31 * 16 * 2);
  float*  XA15 = (float*)alloc((size_t)s15 * 32 * 4);
  float*  XB15 = (float*)alloc((size_t)s15 * 32 * 4);
  float*  H15  = (float*)alloc((size_t)s15 * 32 * 4);
  __bf16* Y15  = (__bf16*)alloc((size_t)s15 * 32 * 2);
  float*  XA7  = (float*)alloc((size_t)s7 * 48 * 4);
  float*  XB7  = (float*)alloc((size_t)s7 * 48 * 4);
  float*  H7   = (float*)alloc((size_t)s7 * 48 * 4);
  __bf16* Y7   = (__bf16*)alloc((size_t)s7 * 48 * 2);
  float*  XA3  = (float*)alloc((size_t)s3 * 96 * 4);
  float*  XB3  = (float*)alloc((size_t)s3 * 96 * 4);
  float*  H3   = (float*)alloc((size_t)s3 * 96 * 4);
  __bf16* Y3   = (__bf16*)alloc((size_t)s3 * 96 * 2);
  float*  XF   = (float*)alloc((size_t)sF * 128 * 4);
  __bf16* YF   = (__bf16*)alloc((size_t)sF * 128 * 2);
  float*  stats = (float*)alloc(257 * 4);
  __bf16* wscr  = (__bf16*)alloc((size_t)524288 * 2);  // largest: lin_w^T (3755x128)

  // ---- helpers (all stream-ordered; graph-capture safe) ----
  auto bn = [&](const float* x, const float* mask, int nSites, int C,
                const float* g, const float* bta, float* y32, __bf16* ybf) {
    k_zero<<<1, 256, 0, stream>>>(stats, 2 * C + 1);
    long total = (long)nSites * C;
    int blocks = (int)((total + 255) / 256);
    if (blocks > 4096) blocks = 4096;
    k_bn_stats<<<blocks, 256, 0, stream>>>(x, mask, stats, nSites, C);
    k_bn_apply<<<(int)((total + 255) / 256), 256, 0, stream>>>(
        x, mask, stats, g, bta, y32, ybf, nSites, C);
  };
  auto pool = [&](const float* mi, float* mo, int Hi, int Wi, int Ho, int Wo, int st) {
    int n = B * Ho * Wo;
    k_pool_mask<<<(n + 255) / 256, 256, 0, stream>>>(mi, mo, B, Hi, Wi, Ho, Wo, st);
  };

  // ---- conv0 (3->16 submanifold) fused with sparse maxpool 3x3/s2: 63 -> 31 ----
  k_conv0_pool<<<(s31 + 255) / 256, 256, 0, stream>>>(features, maskIn, w0, XA31, M31, B);

  // ---- group 0: 16->16, stride 1 @31 ----
  bn(XA31, M31, s31, 16, F(g0p0 + 1), F(g0p0 + 0), XA31, Y31);   // bn0 (ident=XA31)
  run_conv<16, 16, 1, 1>(Y31, F(g0p0 + 4), wscr, M31, nullptr, H31, B, 31, 31, 31, 31, stream);
  bn(H31, M31, s31, 16, F(g0p0 + 3), F(g0p0 + 2), nullptr, Y31); // bn1
  run_conv<16, 16, 1, 1>(Y31, F(g0p0 + 5), wscr, M31, XA31, XB31, B, 31, 31, 31, 31, stream);
  bn(XB31, M31, s31, 16, F(g0p1 + 1), F(g0p1 + 0), nullptr, Y31);
  run_conv<16, 16, 1, 1>(Y31, F(g0p1 + 4), wscr, M31, nullptr, H31, B, 31, 31, 31, 31, stream);
  bn(H31, M31, s31, 16, F(g0p1 + 3), F(g0p1 + 2), nullptr, Y31);
  run_conv<16, 16, 1, 1>(Y31, F(g0p1 + 5), wscr, M31, XB31, XA31, B, 31, 31, 31, 31, stream);

  // ---- group 1: 16->32, stride 2: 31 -> 15 ----
  bn(XA31, M31, s31, 16, F(g1p0 + 1), F(g1p0 + 0), nullptr, Y31);
  pool(M31, M15, 31, 31, 15, 15, 2);
  run_conv<16, 32, 2, 0>(Y31, F(g1p0 + 6), wscr, M15, nullptr, XB15, B, 31, 31, 15, 15, stream); // wid
  run_conv<16, 32, 2, 0>(Y31, F(g1p0 + 4), wscr, M15, nullptr, H15,  B, 31, 31, 15, 15, stream); // wa
  bn(H15, M15, s15, 32, F(g1p0 + 3), F(g1p0 + 2), nullptr, Y15);
  run_conv<32, 32, 1, 1>(Y15, F(g1p0 + 5), wscr, M15, XB15, XA15, B, 15, 15, 15, 15, stream);
  bn(XA15, M15, s15, 32, F(g1p1 + 1), F(g1p1 + 0), nullptr, Y15);
  run_conv<32, 32, 1, 1>(Y15, F(g1p1 + 4), wscr, M15, nullptr, H15, B, 15, 15, 15, 15, stream);
  bn(H15, M15, s15, 32, F(g1p1 + 3), F(g1p1 + 2), nullptr, Y15);
  run_conv<32, 32, 1, 1>(Y15, F(g1p1 + 5), wscr, M15, XA15, XB15, B, 15, 15, 15, 15, stream);

  // ---- group 2: 32->48, stride 2: 15 -> 7 ----
  bn(XB15, M15, s15, 32, F(g2p0 + 1), F(g2p0 + 0), nullptr, Y15);
  pool(M15, M7, 15, 15, 7, 7, 2);
  run_conv<32, 48, 2, 0>(Y15, F(g2p0 + 6), wscr, M7, nullptr, XA7, B, 15, 15, 7, 7, stream);
  run_conv<32, 48, 2, 0>(Y15, F(g2p0 + 4), wscr, M7, nullptr, H7,  B, 15, 15, 7, 7, stream);
  bn(H7, M7, s7, 48, F(g2p0 + 3), F(g2p0 + 2), nullptr, Y7);
  run_conv<48, 48, 1, 1>(Y7, F(g2p0 + 5), wscr, M7, XA7, XB7, B, 7, 7, 7, 7, stream);
  bn(XB7, M7, s7, 48, F(g2p1 + 1), F(g2p1 + 0), nullptr, Y7);
  run_conv<48, 48, 1, 1>(Y7, F(g2p1 + 4), wscr, M7, nullptr, H7, B, 7, 7, 7, 7, stream);
  bn(H7, M7, s7, 48, F(g2p1 + 3), F(g2p1 + 2), nullptr, Y7);
  run_conv<48, 48, 1, 1>(Y7, F(g2p1 + 5), wscr, M7, XB7, XA7, B, 7, 7, 7, 7, stream);

  // ---- group 3: 48->96, stride 2: 7 -> 3 ----
  bn(XA7, M7, s7, 48, F(g3p0 + 1), F(g3p0 + 0), nullptr, Y7);
  pool(M7, M3, 7, 7, 3, 3, 2);
  run_conv<48, 96, 2, 0>(Y7, F(g3p0 + 6), wscr, M3, nullptr, XA3, B, 7, 7, 3, 3, stream);
  run_conv<48, 96, 2, 0>(Y7, F(g3p0 + 4), wscr, M3, nullptr, H3,  B, 7, 7, 3, 3, stream);
  bn(H3, M3, s3, 96, F(g3p0 + 3), F(g3p0 + 2), nullptr, Y3);
  run_conv<96, 96, 1, 1>(Y3, F(g3p0 + 5), wscr, M3, XA3, XB3, B, 3, 3, 3, 3, stream);
  bn(XB3, M3, s3, 96, F(g3p1 + 1), F(g3p1 + 0), nullptr, Y3);
  run_conv<96, 96, 1, 1>(Y3, F(g3p1 + 4), wscr, M3, nullptr, H3, B, 3, 3, 3, 3, stream);
  bn(H3, M3, s3, 96, F(g3p1 + 3), F(g3p1 + 2), nullptr, Y3);
  run_conv<96, 96, 1, 1>(Y3, F(g3p1 + 5), wscr, M3, XB3, XA3, B, 3, 3, 3, 3, stream);

  // ---- tail: BN, 3x3 VALID conv 96->128 (->1x1), BN, linear 128->3755 ----
  bn(XA3, M3, s3, 96, bn_last_g, bn_last_b, nullptr, Y3);
  pool(M3, MF, 3, 3, 1, 1, 1);
  run_conv<96, 128, 1, 0>(Y3, wf, wscr, MF, nullptr, XF, B, 3, 3, 1, 1, stream);
  bn(XF, MF, sF, 128, bnf_g, bnf_b, nullptr, YF);
  k_cvt_wT<<<(128 * 3755 + 255) / 256, 256, 0, stream>>>(lin_w, wscr, 128, 3755);
  dim3 gg((64 + 3) / 4, (3755 + 15) / 16);
  dim3 bb(32, 4);
  k_gemm_wmma<<<gg, bb, 0, stream>>>(YF, wscr, lin_b, (float*)d_out, 1024, 128, 3755);
}
